// ImprovedGraphSAGE_67095979099095
// MI455X (gfx1250) — matrix operand order repaired and checked by reading, not compiled, not run on hardware
//
#include <hip/hip_runtime.h>

#define N_FEAT 165
#define KPAD0  192      // 165 padded up to a multiple of 32 (WMMA K-chunk)
#define HDIM   128
#define LN_EPS 1e-5f

typedef __attribute__((ext_vector_type(16))) __bf16 v16bf;
typedef __attribute__((ext_vector_type(8)))  __bf16 v8bf;
typedef __attribute__((ext_vector_type(8)))  float  v8f;

// ---------------- utility kernels ----------------

__global__ void zero_f32_kernel(float* __restrict__ p, long long n) {
    long long i = (long long)blockIdx.x * blockDim.x + threadIdx.x;
    long long stride = (long long)gridDim.x * blockDim.x;
    for (; i < n; i += stride) p[i] = 0.0f;
}

__global__ void degree_kernel(const int* __restrict__ dst, float* __restrict__ cnt, int E) {
    int i = blockIdx.x * blockDim.x + threadIdx.x;
    int stride = gridDim.x * blockDim.x;
    for (; i < E; i += stride) atomicAdd(&cnt[dst[i]], 1.0f);
}

// layer-0 scatter: one wave per edge, lanes stride the 165 features (coalesced)
__global__ void scatter0_kernel(const float* __restrict__ x,
                                const int* __restrict__ src, const int* __restrict__ dst,
                                float* __restrict__ agg, int E) {
    int wid  = (blockIdx.x * blockDim.x + threadIdx.x) >> 5;
    int lane = threadIdx.x & 31;
    if (wid >= E) return;
    int s = src[wid], d = dst[wid];
    const float* xs = x + (size_t)s * N_FEAT;
    float* ag = agg + (size_t)d * KPAD0;
    for (int k = lane; k < N_FEAT; k += 32)
        atomicAdd(&ag[k], xs[k]);
}

// hidden-layer scatter: one wave per edge, each lane moves 4 contiguous floats
__global__ void scatterH_kernel(const float* __restrict__ h,
                                const int* __restrict__ src, const int* __restrict__ dst,
                                float* __restrict__ agg, int E) {
    int wid  = (blockIdx.x * blockDim.x + threadIdx.x) >> 5;
    int lane = threadIdx.x & 31;
    if (wid >= E) return;
    int s = src[wid], d = dst[wid];
    float4 v = ((const float4*)(h + (size_t)s * HDIM))[lane];
    float* ag = agg + (size_t)d * HDIM + lane * 4;
    atomicAdd(ag + 0, v.x);
    atomicAdd(ag + 1, v.y);
    atomicAdd(ag + 2, v.z);
    atomicAdd(ag + 3, v.w);
}

// mean = agg / max(cnt,1) -> bf16 ; h (source dim Khsrc) -> bf16 padded to Kpad
__global__ void prep_kernel(const float* __restrict__ agg, const float* __restrict__ cnt,
                            const float* __restrict__ hsrc,
                            __bf16* __restrict__ meanb, __bf16* __restrict__ hb,
                            int N, int Kpad, int Khsrc) {
    long long n = (long long)N * Kpad;
    long long i = (long long)blockIdx.x * blockDim.x + threadIdx.x;
    long long stride = (long long)gridDim.x * blockDim.x;
    for (; i < n; i += stride) {
        int row = (int)(i / Kpad);
        int k   = (int)(i - (long long)row * Kpad);
        float inv = 1.0f / fmaxf(cnt[row], 1.0f);
        meanb[i] = (__bf16)(agg[i] * inv);
        float hv = (k < Khsrc) ? hsrc[(size_t)row * Khsrc + k] : 0.0f;
        hb[i] = (__bf16)hv;
    }
}

// fp32 weight [rows][Ksrc] -> bf16 [rows][Kpad] zero-padded
__global__ void convw_kernel(const float* __restrict__ W, __bf16* __restrict__ Wb,
                             int rows, int Ksrc, int Kpad) {
    long long n = (long long)rows * Kpad;
    long long i = (long long)blockIdx.x * blockDim.x + threadIdx.x;
    long long stride = (long long)gridDim.x * blockDim.x;
    for (; i < n; i += stride) {
        int r = (int)(i / Kpad);
        int k = (int)(i - (long long)r * Kpad);
        Wb[i] = (k < Ksrc) ? (__bf16)W[(size_t)r * Ksrc + k] : (__bf16)0.0f;
    }
}

// ---------------- WMMA fragment loaders (ISA 7.12.2 layouts, wave32) ----------------

// A 16x32 bf16: lane m=lane%16, g=lane/16; halves: K = (t<8?0:16) + 8g + (t&7)
__device__ __forceinline__ v16bf load_a_frag(const __bf16* __restrict__ A, int K,
                                             int row, int k0, int g) {
    const __bf16* base = A + (size_t)row * K + k0 + 8 * g;
    v8bf lo = *(const v8bf*)(base);        // K = k0+8g   .. +8
    v8bf hi = *(const v8bf*)(base + 16);   // K = k0+16+8g.. +8
    v16bf f;
#pragma unroll
    for (int i = 0; i < 8; ++i) { f[i] = lo[i]; f[8 + i] = hi[i]; }
    return f;
}

// B 32x16 bf16 (B = W^T, W row-major [out][K]): lane n=lane%16; halves K = (t<8?0:16) + 8g + (t&7)
__device__ __forceinline__ v16bf load_b_frag(const __bf16* __restrict__ W, int K,
                                             int col, int k0, int g) {
    const __bf16* base = W + (size_t)col * K + k0 + 8 * g;
    v8bf lo = *(const v8bf*)(base);
    v8bf hi = *(const v8bf*)(base + 16);
    v16bf f;
#pragma unroll
    for (int i = 0; i < 8; ++i) { f[i] = lo[i]; f[8 + i] = hi[i]; }
    return f;
}

// ---------------- fused SAGE combine: out = epilogue(mean@W1^T + b + h@W2^T) ----------------
// block = 256 threads (8 waves); block computes a 16-node x 128-col tile; wave w -> cols [16w,16w+16)
// mode 0: ReLU only (layer 0).  mode 1: + residual, LayerNorm, ReLU (layers 1..2).

__global__ __launch_bounds__(256)
void sage_gemm_kernel(const __bf16* __restrict__ A1, const __bf16* __restrict__ W1,
                      const __bf16* __restrict__ A2, const __bf16* __restrict__ W2,
                      const float* __restrict__ bias, const float* __restrict__ resid,
                      float* __restrict__ out, int N, int K, int mode) {
    __shared__ float tile[16][HDIM];
    __shared__ float redS[16][17];
    __shared__ float redQ[16][17];

    const int wave = threadIdx.x >> 5;
    const int lane = threadIdx.x & 31;
    const int g    = lane >> 4;
    const int ln   = lane & 15;
    const int nodeBase = blockIdx.x * 16;

    int arow = nodeBase + ln;
    if (arow >= N) arow = N - 1;              // clamp loads; stores are guarded
    const int ncol = wave * 16 + ln;

    v8f c = {};
    for (int k0 = 0; k0 < K; k0 += 32) {      // mean @ Wl^T
        v16bf a = load_a_frag(A1, K, arow, k0, g);
        v16bf b = load_b_frag(W1, K, ncol, k0, g);
        c = __builtin_amdgcn_wmma_f32_16x16x32_bf16(false, a, false, b, (short)0, c, false, false);
    }
    for (int k0 = 0; k0 < K; k0 += 32) {      // h @ Wr^T (same K: padded identically)
        v16bf a = load_a_frag(A2, K, arow, k0, g);
        v16bf b = load_b_frag(W2, K, ncol, k0, g);
        c = __builtin_amdgcn_wmma_f32_16x16x32_bf16(false, a, false, b, (short)0, c, false, false);
    }

    const float bn = bias[ncol];

    if (mode == 0) {
#pragma unroll
        for (int r = 0; r < 8; ++r) {
            int row = r + 8 * g;              // C layout: VGPR r -> M = r + 8*(lane/16)
            int node = nodeBase + row;
            if (node < N) {
                float v = c[r] + bn;
                out[(size_t)node * HDIM + ncol] = v > 0.0f ? v : 0.0f;
            }
        }
    } else {
#pragma unroll
        for (int r = 0; r < 8; ++r) {
            int row = r + 8 * g;
            int node = nodeBase + row;
            int rnode = node < N ? node : N - 1;
            tile[row][ncol] = c[r] + bn + resid[(size_t)rnode * HDIM + ncol];
        }
        __syncthreads();

        const int row = threadIdx.x >> 4;     // 16 rows x 16 partial-sum threads
        const int sub = threadIdx.x & 15;
        float s = 0.0f, q = 0.0f;
#pragma unroll
        for (int j = 0; j < 8; ++j) {
            float v = tile[row][sub * 8 + j];
            s += v; q += v * v;
        }
        redS[row][sub] = s; redQ[row][sub] = q;
        __syncthreads();
        if (sub == 0) {
            float S = 0.0f, Q = 0.0f;
#pragma unroll
            for (int t = 0; t < 16; ++t) { S += redS[row][t]; Q += redQ[row][t]; }
            redS[row][16] = S; redQ[row][16] = Q;
        }
        __syncthreads();
        const float mu  = redS[row][16] * (1.0f / HDIM);
        const float var = redQ[row][16] * (1.0f / HDIM) - mu * mu;
        const float inv = rsqrtf(var + LN_EPS);
        const int node  = nodeBase + row;
        if (node < N) {
#pragma unroll
            for (int j = 0; j < 8; ++j) {
                int col = sub * 8 + j;
                float v = (tile[row][col] - mu) * inv;
                out[(size_t)node * HDIM + col] = v > 0.0f ? v : 0.0f;
            }
        }
    }
}

// ---------------- head: out[N,2] = h @ lin_W^T + lin_b (wave per node) ----------------

__global__ void head_kernel(const float* __restrict__ h, const float* __restrict__ linW,
                            const float* __restrict__ linb, float* __restrict__ out, int N) {
    int wid  = (blockIdx.x * blockDim.x + threadIdx.x) >> 5;
    int lane = threadIdx.x & 31;
    if (wid >= N) return;
    float4 v  = ((const float4*)(h + (size_t)wid * HDIM))[lane];
    float4 w0 = ((const float4*)(linW))[lane];
    float4 w1 = ((const float4*)(linW + HDIM))[lane];
    float p0 = v.x * w0.x + v.y * w0.y + v.z * w0.z + v.w * w0.w;
    float p1 = v.x * w1.x + v.y * w1.y + v.z * w1.z + v.w * w1.w;
#pragma unroll
    for (int off = 16; off > 0; off >>= 1) {
        p0 += __shfl_xor(p0, off, 32);
        p1 += __shfl_xor(p1, off, 32);
    }
    if (lane == 0) {
        out[(size_t)wid * 2 + 0] = p0 + linb[0];
        out[(size_t)wid * 2 + 1] = p1 + linb[1];
    }
}

// ---------------- launcher ----------------

extern "C" void kernel_launch(void* const* d_in, const int* in_sizes, int n_in,
                              void* d_out, int out_size, void* d_ws, size_t ws_size,
                              hipStream_t stream) {
    const float* x    = (const float*)d_in[0];
    const int*   ei   = (const int*)d_in[1];
    const float* Wl0  = (const float*)d_in[2];
    const float* bl0  = (const float*)d_in[3];
    const float* Wr0  = (const float*)d_in[4];
    const float* Wl   = (const float*)d_in[5];   // [2][H][H]
    const float* bl   = (const float*)d_in[6];   // [2][H]
    const float* Wr   = (const float*)d_in[7];   // [2][H][H]
    const float* linW = (const float*)d_in[8];   // [2][H]
    const float* linb = (const float*)d_in[9];   // [2]
    float* out = (float*)d_out;

    const int N = in_sizes[0] / N_FEAT;
    const int E = in_sizes[1] / 2;
    const int* src = ei;
    const int* dst = ei + E;

    // ---- workspace carve (256B aligned) ----
    char* p = (char*)d_ws;
    auto alloc = [&](size_t bytes) -> char* {
        char* r = p; p += (bytes + 255) & ~(size_t)255; return r;
    };
    float*  cnt   = (float*) alloc((size_t)N * 4);
    float*  agg   = (float*) alloc((size_t)N * KPAD0 * 4);
    __bf16* meanb = (__bf16*)alloc((size_t)N * KPAD0 * 2);
    __bf16* hb    = (__bf16*)alloc((size_t)N * KPAD0 * 2);
    float*  hA    = (float*) alloc((size_t)N * HDIM * 4);
    float*  hB    = (float*) alloc((size_t)N * HDIM * 4);
    __bf16* Wl0b  = (__bf16*)alloc((size_t)HDIM * KPAD0 * 2);
    __bf16* Wr0b  = (__bf16*)alloc((size_t)HDIM * KPAD0 * 2);
    __bf16* Wl1b  = (__bf16*)alloc((size_t)HDIM * HDIM * 2);
    __bf16* Wr1b  = (__bf16*)alloc((size_t)HDIM * HDIM * 2);
    __bf16* Wl2b  = (__bf16*)alloc((size_t)HDIM * HDIM * 2);
    __bf16* Wr2b  = (__bf16*)alloc((size_t)HDIM * HDIM * 2);

    const int TB = 256;
    auto cdiv = [](long long a, long long b) { return (int)((a + b - 1) / b); };

    // ---- weights -> bf16 (padded) ----
    convw_kernel<<<cdiv((long long)HDIM * KPAD0, TB), TB, 0, stream>>>(Wl0, Wl0b, HDIM, N_FEAT, KPAD0);
    convw_kernel<<<cdiv((long long)HDIM * KPAD0, TB), TB, 0, stream>>>(Wr0, Wr0b, HDIM, N_FEAT, KPAD0);
    convw_kernel<<<cdiv((long long)HDIM * HDIM, TB), TB, 0, stream>>>(Wl,              Wl1b, HDIM, HDIM, HDIM);
    convw_kernel<<<cdiv((long long)HDIM * HDIM, TB), TB, 0, stream>>>(Wr,              Wr1b, HDIM, HDIM, HDIM);
    convw_kernel<<<cdiv((long long)HDIM * HDIM, TB), TB, 0, stream>>>(Wl + HDIM * HDIM, Wl2b, HDIM, HDIM, HDIM);
    convw_kernel<<<cdiv((long long)HDIM * HDIM, TB), TB, 0, stream>>>(Wr + HDIM * HDIM, Wr2b, HDIM, HDIM, HDIM);

    // ---- degrees (shared by all layers) ----
    zero_f32_kernel<<<cdiv(N, TB), TB, 0, stream>>>(cnt, N);
    degree_kernel<<<cdiv(E, TB), TB, 0, stream>>>(dst, cnt, E);

    const int scatterBlocks = cdiv((long long)E * 32, TB);
    const int gemmBlocks    = cdiv(N, 16);

    // ---- layer 0: d=165 (pad 192) -> hA ----
    zero_f32_kernel<<<cdiv((long long)N * KPAD0, TB), TB, 0, stream>>>(agg, (long long)N * KPAD0);
    scatter0_kernel<<<scatterBlocks, TB, 0, stream>>>(x, src, dst, agg, E);
    prep_kernel<<<cdiv((long long)N * KPAD0, TB), TB, 0, stream>>>(agg, cnt, x, meanb, hb, N, KPAD0, N_FEAT);
    sage_gemm_kernel<<<gemmBlocks, TB, 0, stream>>>(meanb, Wl0b, hb, Wr0b, bl0, nullptr, hA, N, KPAD0, 0);

    // ---- layer 1: hA -> hB (residual + LN + ReLU) ----
    zero_f32_kernel<<<cdiv((long long)N * HDIM, TB), TB, 0, stream>>>(agg, (long long)N * HDIM);
    scatterH_kernel<<<scatterBlocks, TB, 0, stream>>>(hA, src, dst, agg, E);
    prep_kernel<<<cdiv((long long)N * HDIM, TB), TB, 0, stream>>>(agg, cnt, hA, meanb, hb, N, HDIM, HDIM);
    sage_gemm_kernel<<<gemmBlocks, TB, 0, stream>>>(meanb, Wl1b, hb, Wr1b, bl, hA, hB, N, HDIM, 1);

    // ---- layer 2: hB -> hA ----
    zero_f32_kernel<<<cdiv((long long)N * HDIM, TB), TB, 0, stream>>>(agg, (long long)N * HDIM);
    scatterH_kernel<<<scatterBlocks, TB, 0, stream>>>(hB, src, dst, agg, E);
    prep_kernel<<<cdiv((long long)N * HDIM, TB), TB, 0, stream>>>(agg, cnt, hB, meanb, hb, N, HDIM, HDIM);
    sage_gemm_kernel<<<gemmBlocks, TB, 0, stream>>>(meanb, Wl2b, hb, Wr2b, bl + HDIM, hB, hA, N, HDIM, 1);

    // ---- head ----
    head_kernel<<<cdiv((long long)N * 32, TB), TB, 0, stream>>>(hA, linW, linb, out, N);
}
